// AttentionOperator_4776003633163
// MI455X (gfx1250) — matrix operand order repaired and checked by reading, not compiled, not run on hardware
//
#include <hip/hip_runtime.h>

typedef __attribute__((ext_vector_type(16))) _Float16 v16h;
typedef __attribute__((ext_vector_type(8)))  _Float16 v8h;
typedef __attribute__((ext_vector_type(4)))  _Float16 v4h;
typedef __attribute__((ext_vector_type(8)))  float    v8f;
typedef __attribute__((ext_vector_type(4)))  float    v4f;

#define CLAMP_V 1e4f

__device__ __forceinline__ float clipf(float x) {
  return fminf(fmaxf(x, -CLAMP_V), CLAMP_V);
}

// ---------------- WMMA fragment helpers (wave32, 16x16x32 f16) ----------------
// A fragment (16x32) from row-major M: two ds_load_b128 per lane.
__device__ __forceinline__ v16h load_a16(const _Float16* M, int ld, int r0, int k0) {
  int lane = threadIdx.x & 31;
  int m = lane & 15, hi = lane >> 4;
  const _Float16* p = M + (r0 + m) * ld + k0 + hi * 8;
  union { v16h v; v8h h[2]; } u;
  u.h[0] = *(const v8h*)p;
  u.h[1] = *(const v8h*)(p + 16);
  return u.v;
}

// B fragment (32x16) where B = X^T and X is stored row-major [N][K]:
// B[k][n] = X[n][k]  -> contiguous along K: two ds_load_b128 per lane.
__device__ __forceinline__ v16h load_bT16(const _Float16* X, int ld, int k0, int n0) {
  int lane = threadIdx.x & 31;
  int n = lane & 15, hi = lane >> 4;
  const _Float16* p = X + (n0 + n) * ld + k0 + hi * 8;
  union { v16h v; v8h h[2]; } u;
  u.h[0] = *(const v8h*)p;
  u.h[1] = *(const v8h*)(p + 16);
  return u.v;
}

__device__ __forceinline__ v8f wmma_f16(v16h a, v16h b, v8f c) {
  return __builtin_amdgcn_wmma_f32_16x16x32_f16(
      /*neg_a=*/false, a, /*neg_b=*/false, b,
      /*c_mod=*/(short)0, c, /*reuse_a=*/false, /*reuse_b=*/false);
}

// -----------------------------------------------------------------------------
// One workgroup = one (b,h) stream x one value-column group (DV=32 of 128).
// 256 threads = 8 wave32. f32 state tiles live in per-wave VGPR accumulators;
// only an f16 transposed shadow (for B-operand use) lives in LDS.
// -----------------------------------------------------------------------------
#define BD 128   // key/feature dim D
#define BC 64    // chunk size C
#define DV 32    // value columns per workgroup
#define VG 4     // value groups (BD / DV)
#define NCH 64   // chunks (S / C)
#define SEQ 4096

// LDS layout (bytes), all f16
#define OFF_STATET16 0        // [32][128]  state^T   8192
#define OFF_Q        8192     // [64][128]           16384
#define OFF_K        24576    // [64][128]           16384
#define OFF_KT       40960    // [128][64]  k^T      16384
#define OFF_KB       57344    // [64][128]  k*beta   16384
#define OFF_KBT      73728    // [128][64]  (k*b)^T  16384
#define OFF_VBT      90112    // [32][64]   (v*b)^T   4096
#define OFF_T        94208    // [64][64]             8192
#define OFF_A        102400   // [64][64]             8192
#define OFF_W        110592   // [64][128]  -w       16384
#define OFF_UIT      126976   // [32][64]   u_i^T     4096
#define LDS_BYTES    131072   // 128 KB -> 2 workgroups per 320 KB WGP

__global__ void __launch_bounds__(256)
delta_rule_cdna5(const float* __restrict__ qg, const float* __restrict__ kg,
                 const float* __restrict__ vg_, const float* __restrict__ bg,
                 float* __restrict__ outg, float* __restrict__ out_state) {
  extern __shared__ __align__(16) char smem[];
  _Float16* sStateT = (_Float16*)(smem + OFF_STATET16);
  _Float16* sQ      = (_Float16*)(smem + OFF_Q);
  _Float16* sK      = (_Float16*)(smem + OFF_K);
  _Float16* sKT     = (_Float16*)(smem + OFF_KT);
  _Float16* sKB     = (_Float16*)(smem + OFF_KB);
  _Float16* sKBt    = (_Float16*)(smem + OFF_KBT);
  _Float16* sVBt    = (_Float16*)(smem + OFF_VBT);
  _Float16* sT      = (_Float16*)(smem + OFF_T);
  _Float16* sA      = (_Float16*)(smem + OFF_A);
  _Float16* sW      = (_Float16*)(smem + OFF_W);
  _Float16* sUIt    = (_Float16*)(smem + OFF_UIT);

  const int bh   = blockIdx.x / VG;   // (b*H+h) stream
  const int vgrp = blockIdx.x % VG;   // value-column group
  const int tid  = threadIdx.x;
  const int wave = tid >> 5;
  const int lane = tid & 31;
  const int hi   = lane >> 4;
  const int ln   = lane & 15;
  const int vcol0 = vgrp * DV;

  const size_t base = (size_t)bh * SEQ * BD;
  const float* gq = qg  + base;
  const float* gk = kg  + base;
  const float* gv = vg_ + base;
  const float* gb = bg  + base;
  float*       go = outg + base;

  // Per-wave resident f32 state tiles: rows [wave*16, wave*16+16), cols i*16..
  v8f stateAcc[2] = {};
  for (int i = tid; i < BD * DV; i += 256) sStateT[i] = (_Float16)0.f;
  __syncthreads();

  // Prefetch chunk 0 panels into L2 (global_prefetch_b8): one 128B line/thread
  {
    __builtin_prefetch(gq + tid * 32, 0, 1);
    __builtin_prefetch(gk + tid * 32, 0, 1);
    __builtin_prefetch(gb + tid * 32, 0, 1);
    if (tid < BC) __builtin_prefetch(gv + (size_t)tid * BD + vcol0, 0, 1);
  }

  for (int ch = 0; ch < NCH; ++ch) {
    const size_t coff = (size_t)ch * BC * BD;

    // Prefetch next chunk into L2 while this one is loaded/computed.
    if (ch + 1 < NCH) {
      const size_t cnext = coff + (size_t)BC * BD;
      __builtin_prefetch(gq + cnext + tid * 32, 0, 1);   // 256 thr x 128B = 32KB
      __builtin_prefetch(gk + cnext + tid * 32, 0, 1);
      __builtin_prefetch(gb + cnext + tid * 32, 0, 1);
      if (tid < BC) __builtin_prefetch(gv + cnext + (size_t)tid * BD + vcol0, 0, 1);
    }

    // ---- Phase 0: stage chunk; clip; build row-major + transposed copies ----
#pragma unroll
    for (int j = 0; j < 8; ++j) {            // 64x128 f32: 8 float4 / thread
      int idx = (tid + 256 * j) * 4;
      int row = idx >> 7;                    // c
      int col = idx & 127;                   // d
      v4f q4 = *(const v4f*)(gq + coff + idx);
      v4f k4 = *(const v4f*)(gk + coff + idx);
      v4f b4 = *(const v4f*)(gb + coff + idx);
      v4h q4h, k4h, kb4h;
#pragma unroll
      for (int e = 0; e < 4; ++e) {
        float qc = clipf(q4[e]);
        float kc = clipf(k4[e]);
        float bc = fminf(fmaxf(b4[e], 1e-6f), CLAMP_V);
        _Float16 kh  = (_Float16)kc;
        _Float16 kbh = (_Float16)(kc * bc);
        q4h[e] = (_Float16)qc;
        k4h[e] = kh;
        kb4h[e] = kbh;
        sKT[(col + e) * BC + row]  = kh;     // transposed scatter
        sKBt[(col + e) * BC + row] = kbh;
      }
      *(v4h*)(sQ + idx)  = q4h;              // ds_store_b64
      *(v4h*)(sK + idx)  = k4h;
      *(v4h*)(sKB + idx) = kb4h;
    }
#pragma unroll
    for (int j = 0; j < 2; ++j) {            // 64x32 v-columns of this group
      int idx = (tid + 256 * j) * 4;
      int row = idx >> 5, col = idx & 31;
      size_t goff = coff + (size_t)row * BD + vcol0 + col;
      v4f v4 = *(const v4f*)(gv + goff);
      v4f b4 = *(const v4f*)(gb + goff);
#pragma unroll
      for (int e = 0; e < 4; ++e) {
        float vc = clipf(v4[e]);
        float bc = fminf(fmaxf(b4[e], 1e-6f), CLAMP_V);
        sVBt[(col + e) * BC + row] = (_Float16)(vc * bc);
      }
    }
    __syncthreads();

    // ---- Phase 1: T = I - stril(kb k^T) ; Aq = tril(q k^T)   [64x64] --------
#pragma unroll
    for (int i = 0; i < 2; ++i) {            // T tiles: [4][4], 2 per wave
      int t = wave * 2 + i, ti = t >> 2, tj = t & 3;
      v8f acc = {};
#pragma unroll
      for (int kk = 0; kk < 4; ++kk)
        acc = wmma_f16(load_a16(sKB, BD, ti * 16, kk * 32),
                       load_bT16(sK, BD, kk * 32, tj * 16), acc);
#pragma unroll
      for (int r = 0; r < 8; ++r) {
        int M = ti * 16 + hi * 8 + r, N = tj * 16 + ln;
        float val = (M == N) ? 1.0f : ((M > N) ? -acc[r] : 0.0f);
        sT[M * BC + N] = (_Float16)val;
      }
    }
#pragma unroll
    for (int i = 0; i < 2; ++i) {            // Aq tiles: [4][4], 2 per wave
      int t = wave * 2 + i, ti = t >> 2, tj = t & 3;
      v8f acc = {};
#pragma unroll
      for (int kk = 0; kk < 4; ++kk)
        acc = wmma_f16(load_a16(sQ, BD, ti * 16, kk * 32),
                       load_bT16(sK, BD, kk * 32, tj * 16), acc);
#pragma unroll
      for (int r = 0; r < 8; ++r) {
        int M = ti * 16 + hi * 8 + r, N = tj * 16 + ln;
        float val = (M >= N) ? acc[r] : 0.0f;
        sA[M * BC + N] = (_Float16)val;
      }
    }
    __syncthreads();

    // ---- Phase 2: -w = -(T@kb) -> LDS ; u = T@vb -> registers ---------------
#pragma unroll
    for (int i = 0; i < 4; ++i) {            // w tiles: [4][8], 4 per wave
      int t = wave * 4 + i, ti = t >> 3, tj = t & 7;
      v8f acc = {};
#pragma unroll
      for (int kk = 0; kk < 2; ++kk)
        acc = wmma_f16(load_a16(sT, BC, ti * 16, kk * 32),
                       load_bT16(sKBt, BC, kk * 32, tj * 16), acc);
#pragma unroll
      for (int r = 0; r < 8; ++r)
        sW[(ti * 16 + hi * 8 + r) * BD + tj * 16 + ln] = (_Float16)(-acc[r]);
    }
    v8f accU = {};                           // u tile: [4][2], 1 per wave
    {
      int ti = wave >> 1, tj = wave & 1;
#pragma unroll
      for (int kk = 0; kk < 2; ++kk)
        accU = wmma_f16(load_a16(sT, BC, ti * 16, kk * 32),
                        load_bT16(sVBt, BC, kk * 32, tj * 16), accU);
    }
    __syncthreads();

    // ---- Phase 3: u_i = u + (-w)@state ; store u_i^T ------------------------
    {
      int ti = wave >> 1, tj = wave & 1;
      v8f acc = accU;
#pragma unroll
      for (int kk = 0; kk < 4; ++kk)
        acc = wmma_f16(load_a16(sW, BD, ti * 16, kk * 32),
                       load_bT16(sStateT, BD, kk * 32, tj * 16), acc);
      v8h h;
#pragma unroll
      for (int r = 0; r < 8; ++r) h[r] = (_Float16)acc[r];
      // u_i^T[col][row]: fixed col per lane, 8 consecutive rows -> one b128
      *(v8h*)(sUIt + (tj * 16 + ln) * BC + ti * 16 + hi * 8) = h;
    }
    __syncthreads();

    // ---- Phase 4: o = q@state + Aq@u_i -> global ----------------------------
    {
      int ti = wave >> 1, tj = wave & 1;
      v8f acc = {};
#pragma unroll
      for (int kk = 0; kk < 4; ++kk)        // q @ state
        acc = wmma_f16(load_a16(sQ, BD, ti * 16, kk * 32),
                       load_bT16(sStateT, BD, kk * 32, tj * 16), acc);
#pragma unroll
      for (int kk = 0; kk < 2; ++kk)        // Aq @ u_i
        acc = wmma_f16(load_a16(sA, BC, ti * 16, kk * 32),
                       load_bT16(sUIt, BC, kk * 32, tj * 16), acc);
      float* gdst = go + coff;
#pragma unroll
      for (int r = 0; r < 8; ++r)
        gdst[(size_t)(ti * 16 + hi * 8 + r) * BD + vcol0 + tj * 16 + ln] = acc[r];
    }
    __syncthreads();   // phase 5 overwrites state^T read above

    // ---- Phase 5: state += k^T @ u_i (register tiles + f16^T shadow) --------
#pragma unroll
    for (int i = 0; i < 2; ++i) {            // [8][2] tiles, 2 per wave
      int ti = wave, tj = i;
      v8f acc = stateAcc[i];
#pragma unroll
      for (int kk = 0; kk < 2; ++kk)
        acc = wmma_f16(load_a16(sKT, BC, ti * 16, kk * 32),
                       load_bT16(sUIt, BC, kk * 32, tj * 16), acc);
      stateAcc[i] = acc;
      v8h h;
#pragma unroll
      for (int r = 0; r < 8; ++r) h[r] = (_Float16)acc[r];
      *(v8h*)(sStateT + (tj * 16 + ln) * BD + ti * 16 + hi * 8) = h;
    }
    __syncthreads();
  }

  // final state -> d_out tail: [B,H,D,D], our DV column slice, from registers
#pragma unroll
  for (int i = 0; i < 2; ++i) {
#pragma unroll
    for (int r = 0; r < 8; ++r) {
      int row = wave * 16 + hi * 8 + r;      // d
      int col = vcol0 + i * 16 + ln;         // e
      out_state[(size_t)bh * BD * BD + (size_t)row * BD + col] = stateAcc[i][r];
    }
  }
}

extern "C" void kernel_launch(void* const* d_in, const int* in_sizes, int n_in,
                              void* d_out, int out_size, void* d_ws, size_t ws_size,
                              hipStream_t stream) {
  (void)in_sizes; (void)n_in; (void)out_size; (void)d_ws; (void)ws_size;
  const float* q    = (const float*)d_in[0];
  const float* k    = (const float*)d_in[1];
  const float* v    = (const float*)d_in[2];
  const float* beta = (const float*)d_in[3];
  // d_in[4] = chunk_size (fixed 64, baked into kernel)

  float* out       = (float*)d_out;
  float* out_state = out + (size_t)2 * 16 * SEQ * BD;   // after [B,H,S,D] output

  hipFuncSetAttribute(reinterpret_cast<const void*>(delta_rule_cdna5),
                      hipFuncAttributeMaxDynamicSharedMemorySize, LDS_BYTES);

  dim3 grid(2 * 16 * VG);   // 128 workgroups
  dim3 block(256);          // 8 wave32
  delta_rule_cdna5<<<grid, block, LDS_BYTES, stream>>>(q, k, v, beta, out, out_state);
}
